// GeoVisGNN_10385230921971
// MI455X (gfx1250) — matrix operand order repaired and checked by reading, not compiled, not run on hardware
//
#include <hip/hip_runtime.h>
#include <hip/hip_bf16.h>
#include <math.h>

// ---------------- types for WMMA ----------------
typedef __bf16 bf16_t;
typedef __attribute__((ext_vector_type(16))) __bf16 v16bf;
typedef __attribute__((ext_vector_type(8)))  __bf16 v8bf;
typedef __attribute__((ext_vector_type(8)))  float  v8f;

union BF16x16 { v16bf v; v8bf h[2]; };

// ---------------- problem constants ----------------
#define Bdim   8
#define Tdim   2048
#define Sn     7
#define Hdim   512
#define Kdim   1024          // 2*H
#define PAIRS  16            // (b,t) pairs per workgroup
#define MROWS  112           // PAIRS * Sn == 7 WMMA M-tiles, zero padding
#define KC     32            // K chunk (bf16 WMMA depth)
#define NCH    (Kdim / KC)   // 32 K chunks
#define NT     16            // N-tiles per wave (256 cols = half of H)
#define THREADS 448          // 14 waves: 7 m-tiles x 2 n-halves

// =====================================================================
// Kernel 1: transpose + bf16 hi/lo split of Wm  [K=1024][N=512] -> [N][K]
// =====================================================================
__global__ void prep_wm_kernel(const float* __restrict__ Wm,
                               bf16_t* __restrict__ Bhi,
                               bf16_t* __restrict__ Blo) {
    int i = blockIdx.x * blockDim.x + threadIdx.x;   // over N*K
    if (i >= Hdim * Kdim) return;
    int n = i >> 10;
    int k = i & 1023;
    float w = Wm[(size_t)k * Hdim + n];
    bf16_t hi = (bf16_t)w;
    Bhi[i] = hi;
    Blo[i] = (bf16_t)(w - (float)hi);
}

// =====================================================================
// Kernel 2: fused messages-GEMM (bf16x3 WMMA, double-buffered LDS,
//           async q prefetch) + masked attention + aggregation
// =====================================================================
__global__ __launch_bounds__(THREADS)
void fused_msg_attn_kernel(const float* __restrict__ query,
                           const float* __restrict__ senders,
                           const float* __restrict__ smask,
                           const float* __restrict__ bm,
                           const bf16_t* __restrict__ Bhi,
                           const bf16_t* __restrict__ Blo,
                           float* __restrict__ out) {
    __shared__ bf16_t sAhi[2][MROWS * KC];     // double-buffered A (hi)
    __shared__ bf16_t sAlo[2][MROWS * KC];     // double-buffered A (lo)
    __shared__ float  sBuf[PAIRS * Hdim];      // phase 1: query, phase 2: agg
    __shared__ float  sLogit[MROWS];
    __shared__ float  sAtt[MROWS];

    const int tid  = threadIdx.x;
    const int wave = tid >> 5;
    const int lane = tid & 31;
    const int m    = wave % 7;        // 16-row M tile
    const int nh   = wave / 7;        // 256-col N half
    const int hl   = lane >> 4;       // lane half (K sub-range select)
    const int ln   = lane & 15;       // N column / M row within tile
    const int p0   = blockIdx.x * PAIRS;
    const size_t aBase = (size_t)blockIdx.x * MROWS * Kdim;  // contiguous slab

    // ---------- async prefetch of the query tile into sBuf (32 KB) ----------
    // Consumed only after the K loop; overlaps the entire GEMM.
    {
        #pragma unroll
        for (int r = 0; r < 5; ++r) {
            int idx = tid + r * THREADS;                  // b128 transfer id
            if (idx < PAIRS * (Hdim / 4)) {
                int p  = idx >> 7;                        // pair 0..15
                int n4 = idx & 127;                       // 16B chunk in row
                const float* g = query + (size_t)(p0 + p) * Hdim + n4 * 4;
                unsigned lds_off = (unsigned)(size_t)&sBuf[idx * 4];
                unsigned long long ga = (unsigned long long)(size_t)g;
                asm volatile("global_load_async_to_lds_b128 %0, %1, off"
                             :: "v"(lds_off), "v"(ga) : "memory");
            }
        }
    }

    v8f c[NT];
    {
        v8f z = {};
        #pragma unroll
        for (int t = 0; t < NT; ++t) c[t] = z;
    }

    const int r_ld  = tid >> 2;       // 0..111 row staged by this thread
    const int q4_ld = tid & 3;        // 8-float chunk within 32
    const int stO   = r_ld * KC + q4_ld * 8;

    // ---------- prologue: stage chunk 0 into buffer 0 ----------
    {
        const float* src = senders + aBase + (size_t)r_ld * Kdim + q4_ld * 8;
        float4 f0 = *(const float4*)(src);
        float4 f1 = *(const float4*)(src + 4);
        float v[8] = {f0.x, f0.y, f0.z, f0.w, f1.x, f1.y, f1.z, f1.w};
        #pragma unroll
        for (int j = 0; j < 8; ++j) {
            bf16_t hi = (bf16_t)v[j];
            sAhi[0][stO + j] = hi;
            sAlo[0][stO + j] = (bf16_t)(v[j] - (float)hi);
        }
    }

    const int arow = 16 * m + ln;
    const int c0   = hl * 16;

    // -------------------- K loop: double-buffered --------------------
    for (int kc = 0; kc < NCH; ++kc) {
        __syncthreads();                       // buf[kc&1] ready; buf[(kc+1)&1] free
        const int cur = kc & 1;

        // early-issue global loads of NEXT chunk (latency hidden by WMMAs)
        float4 f0, f1;
        const bool have_next = (kc + 1) < NCH;
        if (have_next) {
            const float* src = senders + aBase + (size_t)r_ld * Kdim
                             + (kc + 1) * KC + q4_ld * 8;
            f0 = *(const float4*)(src);
            f1 = *(const float4*)(src + 4);
        }
        if (kc + 2 < NCH) {
            __builtin_prefetch(senders + aBase + (size_t)r_ld * Kdim + (kc + 2) * KC, 0, 0);
        }

        // A fragments (16 contiguous K c0..c0+15, matching B)
        BF16x16 ahi, alo;
        ahi.h[0] = *(const v8bf*)&sAhi[cur][arow * KC + c0];
        ahi.h[1] = *(const v8bf*)&sAhi[cur][arow * KC + c0 + 8];
        alo.h[0] = *(const v8bf*)&sAlo[cur][arow * KC + c0];
        alo.h[1] = *(const v8bf*)&sAlo[cur][arow * KC + c0 + 8];

        #pragma unroll
        for (int t = 0; t < NT; ++t) {
            const int nGlob = nh * 256 + t * 16 + ln;    // B column = lane%16
            const size_t boff = (size_t)nGlob * Kdim + kc * KC + hl * 16;
            v16bf bhi = *(const v16bf*)(Bhi + boff);     // 16 contiguous K
            v16bf blo = *(const v16bf*)(Blo + boff);
            c[t] = __builtin_amdgcn_wmma_f32_16x16x32_bf16(
                       false, ahi.v, false, bhi, (short)0, c[t], false, false);
            c[t] = __builtin_amdgcn_wmma_f32_16x16x32_bf16(
                       false, ahi.v, false, blo, (short)0, c[t], false, false);
            c[t] = __builtin_amdgcn_wmma_f32_16x16x32_bf16(
                       false, alo.v, false, bhi, (short)0, c[t], false, false);
        }

        // convert + store next chunk into the other buffer
        if (have_next) {
            const int nxt = (kc + 1) & 1;
            float v[8] = {f0.x, f0.y, f0.z, f0.w, f1.x, f1.y, f1.z, f1.w};
            #pragma unroll
            for (int j = 0; j < 8; ++j) {
                bf16_t hi = (bf16_t)v[j];
                sAhi[nxt][stO + j] = hi;
                sAlo[nxt][stO + j] = (bf16_t)(v[j] - (float)hi);
            }
        }
    }

    // -------------------- bias + ReLU in registers --------------------
    #pragma unroll
    for (int t = 0; t < NT; ++t) {
        const int nGlob = nh * 256 + t * 16 + ln;
        const float b = bm[nGlob];
        #pragma unroll
        for (int v = 0; v < 8; ++v) {
            float x = c[t][v] + b;
            c[t][v] = x > 0.f ? x : 0.f;
        }
    }

    // -------------------- wait for async q tile, zero logits ---------------
#if __has_builtin(__builtin_amdgcn_s_wait_asynccnt)
    __builtin_amdgcn_s_wait_asynccnt(0);
#else
    asm volatile("s_wait_asynccnt 0x0" ::: "memory");
#endif
    if (tid < MROWS) sLogit[tid] = 0.f;
    __syncthreads();

    // The 8 rows this lane holds span exactly two pairs: pA and pA+1.
    const int rowBase  = 16 * m + 8 * hl;        // rows rowBase..rowBase+7
    const int pA       = rowBase / 7;
    const int pB       = pA + 1;
    const int boundary = (pA + 1) * 7 - rowBase; // v < boundary -> pair pA

    // -------------------- logit partials (2 q loads per tile) --------------
    {
        float acc[8] = {0.f, 0.f, 0.f, 0.f, 0.f, 0.f, 0.f, 0.f};
        #pragma unroll
        for (int t = 0; t < NT; ++t) {
            const int n = nh * 256 + t * 16 + ln;
            const float qA = sBuf[pA * Hdim + n];
            const float qB = sBuf[pB * Hdim + n];
            #pragma unroll
            for (int v = 0; v < 8; ++v)
                acc[v] += c[t][v] * (v < boundary ? qA : qB);
        }
        #pragma unroll
        for (int v = 0; v < 8; ++v)
            atomicAdd(&sLogit[rowBase + v], acc[v]);
    }
    __syncthreads();

    // -------------------- masked softmax over S=7 (16 threads) -------------
    if (tid < PAIRS) {
        const float scale = 0.044194173824159216f;   // 1/sqrt(512)
        float l[Sn];
        bool  mk[Sn];
        float mx = -__builtin_inff();
        #pragma unroll
        for (int s = 0; s < Sn; ++s) {
            mk[s] = smask[(size_t)(p0 + tid) * Sn + s] > 0.f;
            l[s]  = mk[s] ? sLogit[tid * Sn + s] * scale : -__builtin_inff();
            mx = fmaxf(mx, l[s]);
        }
        float sum = 0.f, e[Sn];
        #pragma unroll
        for (int s = 0; s < Sn; ++s) {
            e[s] = mk[s] ? __expf(l[s] - mx) : 0.f;
            sum += e[s];
        }
        const float inv = (sum > 0.f) ? 1.f / sum : 0.f;  // fully masked -> 0
        #pragma unroll
        for (int s = 0; s < Sn; ++s) sAtt[tid * Sn + s] = e[s] * inv;
    }
    // zero agg buffer (reuse sBuf) concurrently with softmax (disjoint)
    for (int i = tid; i < PAIRS * Hdim; i += THREADS) sBuf[i] = 0.f;
    __syncthreads();

    // -------------------- weighted aggregation (2 atomics per tile) --------
    {
        float att[8];
        #pragma unroll
        for (int v = 0; v < 8; ++v) att[v] = sAtt[rowBase + v];

        #pragma unroll
        for (int t = 0; t < NT; ++t) {
            const int n = nh * 256 + t * 16 + ln;
            float sumA = 0.f, sumB = 0.f;
            #pragma unroll
            for (int v = 0; v < 8; ++v) {
                const float w = att[v] * c[t][v];
                if (v < boundary) sumA += w; else sumB += w;
            }
            atomicAdd(&sBuf[pA * Hdim + n], sumA);
            atomicAdd(&sBuf[pB * Hdim + n], sumB);
        }
    }
    __syncthreads();

    // -------------------- store agg half of output --------------------
    for (int i = tid; i < PAIRS * Hdim; i += THREADS) {
        int p = i >> 9, n = i & 511;
        out[(size_t)(p0 + p) * (2 * Hdim) + n] = sBuf[i];
    }
}

// =====================================================================
// Kernel 3: backward-fill hx to next update frame; writes out[..., 512:1024]
// =====================================================================
__global__ void backfill_kernel(const float* __restrict__ hx,
                                const int* __restrict__ u,
                                float* __restrict__ out) {
    __shared__ int idx[Tdim];
    const int b = blockIdx.x;
    if (threadIdx.x == 0) {
        int carry = Tdim;
        for (int t = Tdim - 1; t >= 0; --t) {
            if (u[(size_t)b * Tdim + t] > 0) carry = t;
            idx[t] = (carry < Tdim) ? carry : t;
        }
    }
    __syncthreads();
    const int total = Tdim * (Hdim / 4);
    for (int i = threadIdx.x; i < total; i += blockDim.x) {
        int t = i >> 7, n4 = i & 127;
        float4 v = *(const float4*)(hx + ((size_t)b * Tdim + idx[t]) * Hdim + n4 * 4);
        *(float4*)(out + ((size_t)b * Tdim + t) * (2 * Hdim) + Hdim + n4 * 4) = v;
    }
}

// =====================================================================
extern "C" void kernel_launch(void* const* d_in, const int* in_sizes, int n_in,
                              void* d_out, int out_size, void* d_ws, size_t ws_size,
                              hipStream_t stream) {
    const float* query   = (const float*)d_in[0];
    const float* senders = (const float*)d_in[1];
    const float* smask   = (const float*)d_in[2];
    const float* Wm      = (const float*)d_in[3];
    const float* bm      = (const float*)d_in[4];
    const float* hx      = (const float*)d_in[5];
    const int*   u       = (const int*)d_in[6];
    float* out = (float*)d_out;

    bf16_t* Bhi = (bf16_t*)d_ws;                 // 1 MB
    bf16_t* Blo = Bhi + (size_t)Hdim * Kdim;     // +1 MB (needs ws >= 2 MB)

    prep_wm_kernel<<<(Hdim * Kdim + 255) / 256, 256, 0, stream>>>(Wm, Bhi, Blo);

    fused_msg_attn_kernel<<<(Bdim * Tdim) / PAIRS, THREADS, 0, stream>>>(
        query, senders, smask, bm, Bhi, Blo, out);

    backfill_kernel<<<Bdim, 256, 0, stream>>>(hx, u, out);
}
